// PolNetDense_90512140796481
// MI455X (gfx1250) — matrix operand order repaired and checked by reading, not compiled, not run on hardware
//
#include <hip/hip_runtime.h>

#define BB 8
#define NN 1024
#define EE 4096
#define ME (BB*EE)   /* 32768 edge items  */
#define MV (BB*NN)   /* 8192  node items  */

typedef __attribute__((ext_vector_type(16))) _Float16 v16h;
typedef __attribute__((ext_vector_type(8)))  _Float16 v8h;
typedef __attribute__((ext_vector_type(8)))  float    v8f;

// ---------------------------------------------------------------------------
// small elementwise kernels
// ---------------------------------------------------------------------------
__global__ __launch_bounds__(256) void k_fill0(float* p, int n) {
    int i = blockIdx.x * 256 + threadIdx.x;
    if (i < n) p[i] = 0.f;
}

// Extract src/rcv node indices from dense one-hot incidence matrices (B,N,E).
__global__ __launch_bounds__(256) void k_indices(const float* __restrict__ Es,
                                                 const float* __restrict__ Er,
                                                 int* __restrict__ src,
                                                 int* __restrict__ rcv) {
    int i = blockIdx.x * 256 + threadIdx.x;
    if (i >= BB * NN * EE) return;
    int e = i % EE;
    int n = (i / EE) % NN;
    int b = i / (EE * NN);
    if (Es[i] > 0.5f) src[b * EE + e] = n;
    if (Er[i] > 0.5f) rcv[b * EE + e] = n;
}

// Node attribute encoding -> f16, 22 channels padded to stride 32.
__global__ __launch_bounds__(256) void k_encode_va(const int* __restrict__ Va,
                                                   _Float16* __restrict__ Venc) {
    int i = blockIdx.x * 256 + threadIdx.x;
    if (i >= MV) return;
    int b = i / NN, n = i % NN;
    const int* p = Va + (size_t)b * 5 * NN + n;
    int   t0 = p[0] + 1;         // 0..8
    int   t1 = p[NN] + 1;        // 0..4
    float gx = (float)p[2 * NN];
    float gy = (float)p[3 * NN];
    int   t4 = p[4 * NN];        // 0..5
    _Float16* o = Venc + (size_t)i * 32;
    #pragma unroll
    for (int c = 0; c < 32; ++c) o[c] = (_Float16)0.f;
    o[t0]      = (_Float16)1.f;
    o[9 + t1]  = (_Float16)1.f;
    o[14]      = (_Float16)(2.f * gx / 40.f - 1.f);
    o[15]      = (_Float16)(2.f * gy / 40.f - 1.f);
    o[16 + t4] = (_Float16)1.f;
}

// feat_e = [E_in(ie), V[src](iv), V[rcv](iv), u(iu), zero-pad]
__global__ __launch_bounds__(256) void k_feat_e(const int* __restrict__ Ea,
                                                const _Float16* __restrict__ Eprev,
                                                const _Float16* __restrict__ V,
                                                const int* __restrict__ src,
                                                const int* __restrict__ rcv,
                                                const float* __restrict__ u,
                                                _Float16* __restrict__ Xe,
                                                int ie, int iv, int iu, int dinpad, int isblk0) {
    int idx = blockIdx.x * 256 + threadIdx.x;
    if (idx >= ME * dinpad) return;
    int c = idx % dinpad;
    int item = idx / dinpad;
    int b = item / EE;
    float v = 0.f;
    if (c < ie) {
        v = isblk0 ? (Ea[item] == c ? 1.f : 0.f)
                   : (float)Eprev[(size_t)item * 32 + c];
    } else if (c < ie + iv) {
        v = (float)V[((size_t)b * NN + src[item]) * 32 + (c - ie)];
    } else if (c < ie + 2 * iv) {
        v = (float)V[((size_t)b * NN + rcv[item]) * 32 + (c - ie - iv)];
    } else if (c < ie + 2 * iv + iu) {
        v = u[b * 32 + (c - ie - 2 * iv)];
    }
    Xe[idx] = (_Float16)v;
}

// feat_v = [V(iv), agg(oe), u(iu), zero-pad]
__global__ __launch_bounds__(256) void k_feat_v(const _Float16* __restrict__ V,
                                                const float* __restrict__ agg,
                                                const float* __restrict__ u,
                                                _Float16* __restrict__ Xv,
                                                int iv, int oe, int iu, int dinpad) {
    int idx = blockIdx.x * 256 + threadIdx.x;
    if (idx >= MV * dinpad) return;
    int c = idx % dinpad;
    int item = idx / dinpad;
    int b = item / NN;
    float v = 0.f;
    if (c < iv)                v = (float)V[(size_t)item * 32 + c];
    else if (c < iv + oe)      v = agg[(size_t)item * 32 + (c - iv)];
    else if (c < iv + oe + iu) v = u[b * 32 + (c - iv - oe)];
    Xv[idx] = (_Float16)v;
}

// scatter-add out_E by receiver + column sums
__global__ __launch_bounds__(256) void k_edge_post(const _Float16* __restrict__ Ye,
                                                   const int* __restrict__ rcv,
                                                   float* __restrict__ agg,
                                                   float* __restrict__ sumE) {
    int idx = blockIdx.x * 256 + threadIdx.x;
    if (idx >= ME * 32) return;
    int c = idx & 31;
    int item = idx >> 5;
    int b = item / EE;
    float v = (float)Ye[idx];
    atomicAdd(&agg[((size_t)b * NN + rcv[item]) * 32 + c], v);
    atomicAdd(&sumE[b * 32 + c], v);
}

__global__ __launch_bounds__(256) void k_node_post(const _Float16* __restrict__ Yv,
                                                   float* __restrict__ sumV) {
    int idx = blockIdx.x * 256 + threadIdx.x;
    if (idx >= MV * 32) return;
    int c = idx & 31;
    int item = idx >> 5;
    int b = item / NN;
    atomicAdd(&sumV[b * 32 + c], (float)Yv[idx]);
}

// Transpose + f16-convert + zero-pad weights: W(din,dout) -> Wt(dout, din_pad)
__global__ __launch_bounds__(256) void k_prep_w(const float* __restrict__ W,
                                                _Float16* __restrict__ Wt,
                                                int din, int dout, int dinpad) {
    int idx = blockIdx.x * 256 + threadIdx.x;
    if (idx >= dout * dinpad) return;
    int n = idx / dinpad, k = idx % dinpad;
    Wt[idx] = (_Float16)(k < din ? W[(size_t)k * dout + n] : 0.f);
}

// ---------------------------------------------------------------------------
// Fused 3-layer MLP (DINPAD -> 64 relu -> 64 relu -> 32, row mask) via WMMA.
// 128 threads = 4 waves; each wave owns a 16-row slice of a 64-row tile.
// ---------------------------------------------------------------------------
template <int DINPAD>
__global__ __launch_bounds__(128)
void mlp3_wmma(const _Float16* __restrict__ X,
               const _Float16* __restrict__ W1t,
               const _Float16* __restrict__ W2t,
               const _Float16* __restrict__ W3t,
               const float* __restrict__ B1,
               const float* __restrict__ B2,
               const float* __restrict__ B3,
               const int* __restrict__ mask,
               _Float16* __restrict__ Y) {
    __shared__ __attribute__((aligned(16))) _Float16 sX [64 * DINPAD];
    __shared__ __attribute__((aligned(16))) _Float16 sW1[64 * DINPAD];
    __shared__ __attribute__((aligned(16))) _Float16 sW2[64 * 64];
    __shared__ __attribute__((aligned(16))) _Float16 sW3[32 * 64];
    __shared__ __attribute__((aligned(16))) _Float16 sH1[64 * 64];
    __shared__ __attribute__((aligned(16))) _Float16 sH2[64 * 64];
    __shared__ float sB1[64], sB2[64], sB3[32];
    __shared__ int   sM[64];

    const int tid   = threadIdx.x;
    const int item0 = blockIdx.x * 64;

    {   // cooperative 128-bit staged loads
        const v8h* gx = (const v8h*)(X + (size_t)item0 * DINPAD);
        v8h* dx = (v8h*)sX;
        for (int i = tid; i < 64 * DINPAD / 8; i += 128) dx[i] = gx[i];
        const v8h* g1 = (const v8h*)W1t; v8h* d1 = (v8h*)sW1;
        for (int i = tid; i < 64 * DINPAD / 8; i += 128) d1[i] = g1[i];
        const v8h* g2 = (const v8h*)W2t; v8h* d2 = (v8h*)sW2;
        for (int i = tid; i < 64 * 64 / 8; i += 128) d2[i] = g2[i];
        const v8h* g3 = (const v8h*)W3t; v8h* d3 = (v8h*)sW3;
        for (int i = tid; i < 32 * 64 / 8; i += 128) d3[i] = g3[i];
        if (tid < 64)  sB1[tid] = B1[tid];
        if (tid >= 64) sB2[tid - 64] = B2[tid - 64];
        if (tid < 32)  sB3[tid] = B3[tid];
        if (tid < 64)  sM[tid] = mask ? mask[item0 + tid] : 1;
    }
    __syncthreads();

    const int wave = tid >> 5;
    const int lane = tid & 31;
    const int lrow = lane & 15;
    const int lhi  = lane >> 4;
    const int row0 = wave * 16;

    // A fragment: 16x32 f16 tile; lane<16: K 0..15, lane>=16: K 16..31
    auto ldA = [&](const _Float16* base, int stride, int r, int kt) -> v16h {
        const v8h* p = (const v8h*)(base + (size_t)r * stride + kt * 32 + lhi * 16);
        v8h lo = p[0];
        v8h hi = p[1];
        return __builtin_shufflevector(lo, hi, 0, 1, 2, 3, 4, 5, 6, 7,
                                               8, 9, 10, 11, 12, 13, 14, 15);
    };
    // B fragment: 32x16; weights stored transposed [n][k] so 16 K are contiguous
    auto ldB = [&](const _Float16* wt, int stride, int col, int kt) -> v16h {
        const v8h* p = (const v8h*)(wt + (size_t)col * stride + kt * 32 + lhi * 16);
        v8h lo = p[0];
        v8h hi = p[1];
        return __builtin_shufflevector(lo, hi, 0, 1, 2, 3, 4, 5, 6, 7,
                                               8, 9, 10, 11, 12, 13, 14, 15);
    };

    const v8f z = {0.f, 0.f, 0.f, 0.f, 0.f, 0.f, 0.f, 0.f};

    // ---- layer 1: DINPAD -> 64, relu -> sH1
    {
        v8f acc[4] = {z, z, z, z};
        #pragma unroll
        for (int kt = 0; kt < DINPAD / 32; ++kt) {
            v16h a = ldA(sX, DINPAD, row0 + lrow, kt);
            #pragma unroll
            for (int nt = 0; nt < 4; ++nt) {
                v16h b = ldB(sW1, DINPAD, nt * 16 + lrow, kt);
                acc[nt] = __builtin_amdgcn_wmma_f32_16x16x32_f16(
                    false, a, false, b, (short)0, acc[nt], false, false);
            }
        }
        #pragma unroll
        for (int nt = 0; nt < 4; ++nt) {
            int col = nt * 16 + lrow;
            float bias = sB1[col];
            #pragma unroll
            for (int r = 0; r < 8; ++r) {
                float v = acc[nt][r] + bias;
                v = v > 0.f ? v : 0.f;
                sH1[(row0 + r + 8 * lhi) * 64 + col] = (_Float16)v;
            }
        }
    }
    __syncthreads();

    // ---- layer 2: 64 -> 64, relu -> sH2
    {
        v8f acc[4] = {z, z, z, z};
        #pragma unroll
        for (int kt = 0; kt < 2; ++kt) {
            v16h a = ldA(sH1, 64, row0 + lrow, kt);
            #pragma unroll
            for (int nt = 0; nt < 4; ++nt) {
                v16h b = ldB(sW2, 64, nt * 16 + lrow, kt);
                acc[nt] = __builtin_amdgcn_wmma_f32_16x16x32_f16(
                    false, a, false, b, (short)0, acc[nt], false, false);
            }
        }
        #pragma unroll
        for (int nt = 0; nt < 4; ++nt) {
            int col = nt * 16 + lrow;
            float bias = sB2[col];
            #pragma unroll
            for (int r = 0; r < 8; ++r) {
                float v = acc[nt][r] + bias;
                v = v > 0.f ? v : 0.f;
                sH2[(row0 + r + 8 * lhi) * 64 + col] = (_Float16)v;
            }
        }
    }
    __syncthreads();

    // ---- layer 3: 64 -> 32, row-mask, write global
    {
        v8f acc[2] = {z, z};
        #pragma unroll
        for (int kt = 0; kt < 2; ++kt) {
            v16h a = ldA(sH2, 64, row0 + lrow, kt);
            #pragma unroll
            for (int nt = 0; nt < 2; ++nt) {
                v16h b = ldB(sW3, 64, nt * 16 + lrow, kt);
                acc[nt] = __builtin_amdgcn_wmma_f32_16x16x32_f16(
                    false, a, false, b, (short)0, acc[nt], false, false);
            }
        }
        #pragma unroll
        for (int nt = 0; nt < 2; ++nt) {
            int col = nt * 16 + lrow;
            float bias = sB3[col];
            #pragma unroll
            for (int r = 0; r < 8; ++r) {
                int row = row0 + r + 8 * lhi;
                float v = acc[nt][r] + bias;
                if (!sM[row]) v = 0.f;
                Y[(size_t)(item0 + row) * 32 + col] = (_Float16)v;
            }
        }
    }
}

// ---------------------------------------------------------------------------
// tiny per-batch global MLP: P=[u(iu), sumV(32), sumE(oe)] -> 64 -> 64 -> 32
// ---------------------------------------------------------------------------
__global__ __launch_bounds__(64) void k_fu(const float* __restrict__ u,
                                           const float* __restrict__ sumV,
                                           const float* __restrict__ sumE,
                                           const float* __restrict__ W0, const float* __restrict__ b0,
                                           const float* __restrict__ W1, const float* __restrict__ b1,
                                           const float* __restrict__ W2, const float* __restrict__ b2,
                                           int iu, int oe,
                                           float* __restrict__ u_out) {
    __shared__ float x[96], h1[64], h2[64];
    int b = blockIdx.x, t = threadIdx.x;
    int din = iu + 32 + oe;
    for (int c = t; c < din; c += 64) {
        float v;
        if (c < iu)           v = u[b * 32 + c];
        else if (c < iu + 32) v = sumV[b * 32 + (c - iu)];
        else                  v = sumE[b * 32 + (c - iu - 32)];
        x[c] = v;
    }
    __syncthreads();
    float a = b0[t];
    for (int k = 0; k < din; ++k) a += x[k] * W0[(size_t)k * 64 + t];
    h1[t] = a > 0.f ? a : 0.f;
    __syncthreads();
    a = b1[t];
    for (int k = 0; k < 64; ++k) a += h1[k] * W1[(size_t)k * 64 + t];
    h2[t] = a > 0.f ? a : 0.f;
    __syncthreads();
    if (t < 32) {
        a = b2[t];
        for (int k = 0; k < 64; ++k) a += h2[k] * W2[(size_t)k * 32 + t];
        u_out[b * 32 + t] = a;
    }
}

// ---------------------------------------------------------------------------
// output head: feat=[V[last](32), ones(32), u(32)] -> 64 -> 64 -> 50, penalty
// ---------------------------------------------------------------------------
__global__ __launch_bounds__(64) void k_out(const _Float16* __restrict__ Vf,
                                            const float* __restrict__ u,
                                            const int* __restrict__ mask_v,
                                            const int* __restrict__ mask,
                                            const float* __restrict__ W0, const float* __restrict__ b0,
                                            const float* __restrict__ W1, const float* __restrict__ b1,
                                            const float* __restrict__ W2, const float* __restrict__ b2,
                                            float* __restrict__ out) {
    __shared__ float x[96], h1[64], h2[64];
    __shared__ int red[64];
    int b = blockIdx.x, t = threadIdx.x;
    int cnt = 0;
    for (int n = t; n < NN; n += 64) cnt += (mask_v[b * NN + n] != 0);
    red[t] = cnt;
    __syncthreads();
    for (int s = 32; s > 0; s >>= 1) {
        if (t < s) red[t] += red[t + s];
        __syncthreads();
    }
    int last = red[0] - 1;
    for (int c = t; c < 96; c += 64) {
        float v;
        if (c < 32)      v = (float)Vf[((size_t)b * NN + last) * 32 + c];
        else if (c < 64) v = 1.0f;
        else             v = u[b * 32 + (c - 64)];
        x[c] = v;
    }
    __syncthreads();
    float a = b0[t];
    for (int k = 0; k < 96; ++k) a += x[k] * W0[(size_t)k * 64 + t];
    h1[t] = a > 0.f ? a : 0.f;
    __syncthreads();
    a = b1[t];
    for (int k = 0; k < 64; ++k) a += h1[k] * W1[(size_t)k * 64 + t];
    h2[t] = a > 0.f ? a : 0.f;
    __syncthreads();
    if (t < 50) {
        a = b2[t];
        for (int k = 0; k < 64; ++k) a += h2[k] * W2[(size_t)k * 50 + t];
        out[b * 50 + t] = a - (mask[b * 50 + t] ? 0.f : 1.0e10f);
    }
}

// ---------------------------------------------------------------------------
// host side
// ---------------------------------------------------------------------------
static inline int cdiv(int a, int b) { return (a + b - 1) / b; }

static void launch_mlp(int dinpad, int M, const _Float16* X,
                       const _Float16* W1t, const _Float16* W2t, const _Float16* W3t,
                       const float* B1, const float* B2, const float* B3,
                       const int* mask, _Float16* Y, hipStream_t s) {
    dim3 g(M / 64), blk(128);
    if (dinpad == 64)
        mlp3_wmma<64><<<g, blk, 0, s>>>(X, W1t, W2t, W3t, B1, B2, B3, mask, Y);
    else if (dinpad == 96)
        mlp3_wmma<96><<<g, blk, 0, s>>>(X, W1t, W2t, W3t, B1, B2, B3, mask, Y);
    else
        mlp3_wmma<128><<<g, blk, 0, s>>>(X, W1t, W2t, W3t, B1, B2, B3, mask, Y);
}

extern "C" void kernel_launch(void* const* d_in, const int* in_sizes, int n_in,
                              void* d_out, int out_size, void* d_ws, size_t ws_size,
                              hipStream_t stream) {
    (void)in_sizes; (void)n_in; (void)out_size; (void)ws_size;

    const int*   Va     = (const int*)d_in[0];
    const float* Es     = (const float*)d_in[1];
    const float* Er     = (const float*)d_in[2];
    const int*   Ea     = (const int*)d_in[3];
    const int*   mask_e = (const int*)d_in[4];
    const int*   mask_v = (const int*)d_in[5];
    const int*   mask   = (const int*)d_in[6];

    // params flattened: gn[0..3] then out; per block dict keys sorted: fe, fu, fv;
    // each mlp: (W,b) x 3 layers.  mlp index: 0=fe, 1=fu, 2=fv.
    auto Pp = [&](int blk, int mlp, int layer, int wb) -> const float* {
        return (const float*)d_in[7 + blk * 18 + mlp * 6 + layer * 2 + wb];
    };

    char* ws = (char*)d_ws;
    size_t off = 0;
    auto alloc = [&](size_t bytes) -> void* {
        off = (off + 255) & ~(size_t)255;
        void* p = ws + off;
        off += bytes;
        return p;
    };

    int*      srcI = (int*)alloc((size_t)ME * 4);
    int*      rcvI = (int*)alloc((size_t)ME * 4);
    _Float16* Venc = (_Float16*)alloc((size_t)MV * 32 * 2);
    _Float16* V1   = (_Float16*)alloc((size_t)MV * 32 * 2);
    _Float16* V2   = (_Float16*)alloc((size_t)MV * 32 * 2);
    _Float16* Xe   = (_Float16*)alloc((size_t)ME * 128 * 2);
    _Float16* Ye   = (_Float16*)alloc((size_t)ME * 32 * 2);
    _Float16* Xv   = (_Float16*)alloc((size_t)MV * 128 * 2);
    float*    agg  = (float*)alloc((size_t)MV * 32 * 4);
    float*    sumE = (float*)alloc(BB * 32 * 4);
    float*    sumV = (float*)alloc(BB * 32 * 4);
    float*    uBuf = (float*)alloc(BB * 32 * 4);

    // WMMA-path mlps: m 0..2 = fe of blocks 0..2, m 3..6 = fv of blocks 0..3
    const int l1pad[7] = {64, 128, 128, 64, 96, 96, 64};
    const int l1din[7] = {50, 128, 128, 54, 96, 96, 64};
    const int pblk[7]  = {0, 1, 2, 0, 1, 2, 3};
    const int pmlp[7]  = {0, 0, 0, 2, 2, 2, 2};
    _Float16* wt1[7]; _Float16* wt2[7]; _Float16* wt3[7];
    for (int m = 0; m < 7; ++m) {
        wt1[m] = (_Float16*)alloc((size_t)64 * 128 * 2);
        wt2[m] = (_Float16*)alloc((size_t)64 * 64 * 2);
        wt3[m] = (_Float16*)alloc((size_t)32 * 64 * 2);
    }

    // ---- one-time prep (deterministic per launch)
    k_indices<<<cdiv(BB * NN * EE, 256), 256, 0, stream>>>(Es, Er, srcI, rcvI);
    k_encode_va<<<cdiv(MV, 256), 256, 0, stream>>>(Va, Venc);
    for (int m = 0; m < 7; ++m) {
        k_prep_w<<<cdiv(64 * l1pad[m], 256), 256, 0, stream>>>(
            Pp(pblk[m], pmlp[m], 0, 0), wt1[m], l1din[m], 64, l1pad[m]);
        k_prep_w<<<cdiv(64 * 64, 256), 256, 0, stream>>>(
            Pp(pblk[m], pmlp[m], 1, 0), wt2[m], 64, 64, 64);
        k_prep_w<<<cdiv(32 * 64, 256), 256, 0, stream>>>(
            Pp(pblk[m], pmlp[m], 2, 0), wt3[m], 64, 32, 64);
    }

    // ---- per-block dims
    const int ie_[4] = {6, 32, 32, 32};
    const int iv_[4] = {22, 32, 32, 32};
    const int iu_[4] = {0, 32, 32, 32};
    const int oe_[4] = {32, 32, 32, 0};
    const int de_[4] = {64, 128, 128, 0};   // edge-MLP Din_pad (block3 skipped)
    const int dv_[4] = {64, 96, 96, 64};    // node-MLP Din_pad

    const _Float16* Vcur = Venc;
    for (int k = 0; k < 4; ++k) {
        const int ie = ie_[k], iv = iv_[k], iu = iu_[k], oe = oe_[k];

        if (k < 3) {
            // edge phase
            int tot = ME * de_[k];
            k_feat_e<<<cdiv(tot, 256), 256, 0, stream>>>(
                Ea, Ye, Vcur, srcI, rcvI, uBuf, Xe, ie, iv, iu, de_[k], (k == 0) ? 1 : 0);
            launch_mlp(de_[k], ME, Xe, wt1[k], wt2[k], wt3[k],
                       Pp(k, 0, 0, 1), Pp(k, 0, 1, 1), Pp(k, 0, 2, 1),
                       mask_e, Ye, stream);
            k_fill0<<<cdiv(MV * 32, 256), 256, 0, stream>>>(agg, MV * 32);
            k_fill0<<<1, 256, 0, stream>>>(sumE, BB * 32);
            k_edge_post<<<cdiv(ME * 32, 256), 256, 0, stream>>>(Ye, rcvI, agg, sumE);
        }

        // node phase
        int totv = MV * dv_[k];
        k_feat_v<<<cdiv(totv, 256), 256, 0, stream>>>(Vcur, agg, uBuf, Xv, iv, oe, iu, dv_[k]);
        _Float16* Vnext = (k & 1) ? V2 : V1;
        launch_mlp(dv_[k], MV, Xv, wt1[3 + k], wt2[3 + k], wt3[3 + k],
                   Pp(k, 2, 0, 1), Pp(k, 2, 1, 1), Pp(k, 2, 2, 1),
                   mask_v, Vnext, stream);
        k_fill0<<<1, 256, 0, stream>>>(sumV, BB * 32);
        k_node_post<<<cdiv(MV * 32, 256), 256, 0, stream>>>(Vnext, sumV);

        // global phase
        k_fu<<<BB, 64, 0, stream>>>(uBuf, sumV, sumE,
                                    Pp(k, 1, 0, 0), Pp(k, 1, 0, 1),
                                    Pp(k, 1, 1, 0), Pp(k, 1, 1, 1),
                                    Pp(k, 1, 2, 0), Pp(k, 1, 2, 1),
                                    iu, oe, uBuf);
        Vcur = Vnext;
    }

    // ---- output head (out-block fe only; its fv/fu have zero-width outputs)
    k_out<<<BB, 64, 0, stream>>>(Vcur, uBuf, mask_v, mask,
                                 Pp(4, 0, 0, 0), Pp(4, 0, 0, 1),
                                 Pp(4, 0, 1, 0), Pp(4, 0, 1, 1),
                                 Pp(4, 0, 2, 0), Pp(4, 0, 2, 1),
                                 (float*)d_out);
}
